// DIEN_48155173323134
// MI455X (gfx1250) — matrix operand order repaired and checked by reading, not compiled, not run on hardware
//
#include <hip/hip_runtime.h>
#include <hip/hip_bf16.h>
#include <math.h>

// ---------- constants ----------
#define BB 4096
#define TT 200
#define HH 32
#define SIDE 64
#define EPS 1e-3f

typedef __attribute__((ext_vector_type(16))) _Float16 v16h;
typedef __attribute__((ext_vector_type(8)))  float    v8f;
typedef __attribute__((ext_vector_type(4)))  unsigned int v4u;
typedef __attribute__((ext_vector_type(8)))  int      v8i;
typedef __attribute__((ext_vector_type(4)))  int      v4i;

#ifndef __has_builtin
#define __has_builtin(x) 0
#endif
#if __has_builtin(__builtin_amdgcn_tensor_load_to_lds) && __has_builtin(__builtin_amdgcn_s_wait_tensorcnt)
#define HAVE_TDM 1
#else
#define HAVE_TDM 0
#endif

// ---------- helpers ----------
__device__ __forceinline__ float sigmoidf_(float x) { return 1.0f / (1.0f + __expf(-x)); }

__device__ __forceinline__ v8f wmma_f16(v16h a, v16h b, v8f c) {
  return __builtin_amdgcn_wmma_f32_16x16x32_f16(false, a, false, b, (short)0, c, false, false);
}

// A-matrix 16x32 f16 layout (ISA 7.12.2): lane holds row M=lane&15,
// k = (v/4)*16 + (lane/16)*8 + (v&3)*2 + j   for half index i = 2v+j
__device__ __forceinline__ v16h load_a_lds(const float* s, int ld, int kbase, int lane) {
  v16h a;
  const int m   = lane & 15;
  const int khi = (lane >> 4) << 3;
#pragma unroll
  for (int i = 0; i < 16; ++i) {
    int v = i >> 1, j = i & 1;
    int k = ((v >> 2) << 4) + khi + ((v & 3) << 1) + j;
    a[i] = (_Float16)s[m * ld + kbase + k];
  }
  return a;
}

// B-matrix 32x16 f16 layout: lane holds col N=lane&15, k = (i/8)*16 + (lane/16)*8 + (i&7)
__device__ __forceinline__ v16h load_b_global(const float* W, int ldn, int kbase, int nbase,
                                              int K, int N, int lane) {
  v16h b;
  const int n  = lane & 15;
  const int k0 = (lane >> 4) << 3;
#pragma unroll
  for (int i = 0; i < 16; ++i) {
    int k = kbase + k0 + (i & 7) + ((i >> 3) << 4), nn = nbase + n;
    float w = (k < K && nn < N) ? W[(size_t)k * ldn + nn] : 0.0f;
    b[i] = (_Float16)w;
  }
  return b;
}

// ---------- TDM: 2D tile (f32) global -> LDS, rows strided ----------
#if HAVE_TDM
__device__ __forceinline__ void tdm_load_2d(unsigned lds_off, const void* gaddr,
                                            unsigned tile_d0, unsigned tile_d1,
                                            unsigned long long stride0_elems) {
  unsigned long long ga = (unsigned long long)(size_t)gaddr;
  v4u g0 = {0, 0, 0, 0};
  g0[0] = 1u;                                              // count=1 (valid D#)
  g0[1] = lds_off;                                         // lds_addr (bytes)
  g0[2] = (unsigned)(ga & 0xFFFFFFFFu);                    // global_addr[31:0]
  g0[3] = (unsigned)((ga >> 32) & 0x01FFFFFFu) | (2u << 30); // addr[56:32] | type=2
  v8i g1 = {0, 0, 0, 0, 0, 0, 0, 0};
  g1[0] = (int)(2u << 16);                                 // data_size=4B, no flags
  g1[1] = (int)((tile_d0 & 0xFFFFu) << 16);                // tensor_dim0[15:0] (== tile_d0)
  g1[2] = (int)(((tile_d0 >> 16) & 0xFFFFu) | ((tile_d1 & 0xFFFFu) << 16)); // dim0 hi | dim1 lo
  g1[3] = (int)(((tile_d1 >> 16) & 0xFFFFu) | ((tile_d0 & 0xFFFFu) << 16)); // dim1 hi | tile_dim0
  g1[4] = (int)(tile_d1 & 0xFFFFu);                        // tile_dim1 (tile_dim2=0)
  g1[5] = (int)(unsigned)(stride0_elems & 0xFFFFFFFFull);  // dim0_stride[31:0]
  g1[6] = (int)(unsigned)((stride0_elems >> 32) & 0xFFFFull); // dim0_stride[47:32]
  g1[7] = 0;
  v4i z4 = {0, 0, 0, 0};
#if defined(__clang_major__) && (__clang_major__ >= 23)
  v8i z8 = {0, 0, 0, 0, 0, 0, 0, 0};
  __builtin_amdgcn_tensor_load_to_lds(g0, g1, z4, z4, z8, 0);
#else
  __builtin_amdgcn_tensor_load_to_lds(g0, g1, z4, z4, 0);
#endif
}
#endif

// prefetch one timestep's xp block [16 rows x 96 cols f32] into LDS
__device__ __forceinline__ void prefetch_xp(const float* xp, int b0, int t, float* dstLds) {
#if HAVE_TDM
  tdm_load_2d((unsigned)(size_t)dstLds,
              xp + ((size_t)b0 * TT + t) * 96,
              96u, 16u, (unsigned long long)TT * 96ull);
#else
  const int lane = threadIdx.x & 31;
  for (int idx = lane; idx < 16 * 24; idx += 32) {
    int rr = idx / 24, c4 = idx % 24;
    ((float4*)(dstLds + rr * 96))[c4] =
        ((const float4*)(xp + ((size_t)(b0 + rr) * TT + t) * 96))[c4];
  }
#endif
}

__device__ __forceinline__ void wait_xp_1() {
#if HAVE_TDM
  __builtin_amdgcn_s_wait_tensorcnt((short)1);
#else
  __syncthreads();
#endif
}
__device__ __forceinline__ void wait_xp_0() {
#if HAVE_TDM
  __builtin_amdgcn_s_wait_tensorcnt((short)0);
#else
  __syncthreads();
#endif
}

// ---------- kernel 1/5: Y[M,96] = X[M,32] @ W[32,96] + bias ----------
// One wave per 16-row tile: A loaded once, all 6 N-tiles computed (no A re-read).
#define PROJ_WPB 8
__global__ void proj_kernel(const float* __restrict__ X, const float* __restrict__ W,
                            const float* __restrict__ bias, float* __restrict__ Y, int M) {
  __shared__ float sXAll[PROJ_WPB][16][32];
  const int lane = threadIdx.x & 31, wave = threadIdx.x >> 5;
  const int wid  = blockIdx.x * PROJ_WPB + wave;
  const int m0 = wid * 16;
  if (m0 >= M) return;
  const int n = lane & 15, hi8 = (lane >> 4) << 3;
  float (*sX)[32] = sXAll[wave];

  // cooperative b128 stage of the A tile
  for (int idx = lane; idx < 16 * 8; idx += 32) {
    int rr = idx >> 3, c4 = idx & 7;
    ((float4*)&sX[rr][0])[c4] = ((const float4*)(X + (size_t)(m0 + rr) * 32))[c4];
  }
  __syncthreads();
  v16h a = load_a_lds(&sX[0][0], 32, 0, lane);

#pragma unroll
  for (int nt = 0; nt < 6; ++nt) {
    v16h b = load_b_global(W, 96, 0, nt * 16, 32, 96, lane);
    v8f c;
    float bv = bias[nt * 16 + n];
#pragma unroll
    for (int r = 0; r < 8; ++r) c[r] = bv;
    c = wmma_f16(a, b, c);
#pragma unroll
    for (int r = 0; r < 8; ++r)
      Y[(size_t)(m0 + r + hi8) * 96 + nt * 16 + n] = c[r];
  }
}

// ---------- kernel 2: GRU scan over T (16 batch rows per wave) ----------
__global__ void gru_scan_kernel(const float* __restrict__ xp, const unsigned char* __restrict__ mask,
                                const float* __restrict__ Wh, float* __restrict__ ge) {
  __shared__ float sH[16][32];
  __shared__ float sT[16][32];
  __shared__ float sXP[2][16][96];
  const int lane = threadIdx.x & 31;
  const int b0 = blockIdx.x * 16;
  const int n = lane & 15, hi8 = (lane >> 4) << 3;

  // preload recurrent weight B-tiles (gate order z,r,h at cols 0/32/64)
  v16h bz0 = load_b_global(Wh, 96, 0,  0, 32, 96, lane);
  v16h bz1 = load_b_global(Wh, 96, 0, 16, 32, 96, lane);
  v16h br0 = load_b_global(Wh, 96, 0, 32, 32, 96, lane);
  v16h br1 = load_b_global(Wh, 96, 0, 48, 32, 96, lane);
  v16h bh0 = load_b_global(Wh, 96, 0, 64, 32, 96, lane);
  v16h bh1 = load_b_global(Wh, 96, 0, 80, 32, 96, lane);

  v8f h_lo = {}, h_hi = {};
  for (int idx = lane; idx < 16 * 32; idx += 32) sH[idx >> 5][idx & 31] = 0.0f;
  __syncthreads();

  prefetch_xp(xp, b0, 0, &sXP[0][0][0]);            // TENSORcnt -> 1

  for (int t = 0; t < TT; ++t) {
    if (t + 1 < TT) { prefetch_xp(xp, b0, t + 1, &sXP[(t + 1) & 1][0][0]); wait_xp_1(); }
    else            { wait_xp_0(); }
    const float (*xt)[96] = sXP[t & 1];

    v16h ah = load_a_lds(&sH[0][0], 32, 0, lane);
    v8f zero = {};
    v8f hz0 = wmma_f16(ah, bz0, zero);
    v8f hz1 = wmma_f16(ah, bz1, zero);
    v8f hr0 = wmma_f16(ah, br0, zero);
    v8f hr1 = wmma_f16(ah, br1, zero);

    v8f z_lo, z_hi, r_lo, r_hi;
#pragma unroll
    for (int r = 0; r < 8; ++r) {
      const float* xrow = xt[r + hi8];
      z_lo[r] = sigmoidf_(xrow[n]      + hz0[r]);
      z_hi[r] = sigmoidf_(xrow[16 + n] + hz1[r]);
      r_lo[r] = sigmoidf_(xrow[32 + n] + hr0[r]);
      r_hi[r] = sigmoidf_(xrow[48 + n] + hr1[r]);
      sT[r + hi8][n]      = r_lo[r] * h_lo[r];
      sT[r + hi8][16 + n] = r_hi[r] * h_hi[r];
    }
    __syncthreads();

    v16h arh = load_a_lds(&sT[0][0], 32, 0, lane);
    v8f hh0 = wmma_f16(arh, bh0, zero);
    v8f hh1 = wmma_f16(arh, bh1, zero);

#pragma unroll
    for (int r = 0; r < 8; ++r) {
      const int row = b0 + r + hi8;
      const float* xrow = xt[r + hi8];
      float hhl = tanhf(xrow[64 + n] + hh0[r]);
      float hhh = tanhf(xrow[80 + n] + hh1[r]);
      float hn0 = z_lo[r] * h_lo[r] + (1.0f - z_lo[r]) * hhl;
      float hn1 = z_hi[r] * h_hi[r] + (1.0f - z_hi[r]) * hhh;
      bool mk = mask[(size_t)row * TT + t] != 0;
      h_lo[r] = mk ? hn0 : h_lo[r];
      h_hi[r] = mk ? hn1 : h_hi[r];
      ge[((size_t)row * TT + t) * 32 + n]      = h_lo[r];
      ge[((size_t)row * TT + t) * 32 + 16 + n] = h_hi[r];
      sH[r + hi8][n]      = h_lo[r];
      sH[r + hi8][16 + n] = h_hi[r];
    }
    __syncthreads();
  }
}

// ---------- kernel 3: fused DIN attention MLP -> raw scores ----------
// A-operands for the [q,g,q-g,q*g] concat are built in registers from staged q/g.
#define ATT_WPB 2
__global__ void attn_kernel(const float* __restrict__ tgt, const float* __restrict__ ge,
                            const float* __restrict__ W1, const float* __restrict__ b1,
                            const float* __restrict__ W2, const float* __restrict__ b2,
                            const float* __restrict__ W3, const float* __restrict__ b3,
                            float* __restrict__ score) {
  __shared__ float sQAll[ATT_WPB][16][32];
  __shared__ float sGAll[ATT_WPB][16][32];
  __shared__ float sMidAll[ATT_WPB][16][96];
  const int wave = threadIdx.x >> 5, lane = threadIdx.x & 31;
  const int row0 = (blockIdx.x * ATT_WPB + wave) * 16;
  const int n = lane & 15, hi8 = (lane >> 4) << 3;
  float (*sQ)[32]  = sQAll[wave];
  float (*sG)[32]  = sGAll[wave];
  float (*sMid)[96] = sMidAll[wave];

  for (int idx = lane; idx < 16 * 8; idx += 32) {
    int rr = idx >> 3, c4 = idx & 7;
    int grow = row0 + rr;
    int b = grow / TT, t = grow % TT;
    ((float4*)&sG[rr][0])[c4] = ((const float4*)(ge + ((size_t)b * TT + t) * 32))[c4];
    ((float4*)&sQ[rr][0])[c4] = ((const float4*)(tgt + (size_t)b * 32))[c4];
  }
  __syncthreads();

  v16h aq = load_a_lds(&sQ[0][0], 32, 0, lane);
  v16h ag = load_a_lds(&sG[0][0], 32, 0, lane);
  v16h am, ax;
#pragma unroll
  for (int i = 0; i < 16; ++i) { am[i] = aq[i] - ag[i]; ax[i] = aq[i] * ag[i]; }
  v16h aseg[4] = {aq, ag, am, ax};

  // GEMM1: [16,128]@[128,80]+b1 -> sigmoid
  v8f acc[5];
#pragma unroll
  for (int nt = 0; nt < 5; ++nt) {
    float bv = b1[nt * 16 + n];
#pragma unroll
    for (int r = 0; r < 8; ++r) acc[nt][r] = bv;
  }
#pragma unroll
  for (int ks = 0; ks < 4; ++ks) {
#pragma unroll
    for (int nt = 0; nt < 5; ++nt)
      acc[nt] = wmma_f16(aseg[ks], load_b_global(W1, 80, ks * 32, nt * 16, 128, 80, lane), acc[nt]);
  }
#pragma unroll
  for (int nt = 0; nt < 5; ++nt)
#pragma unroll
    for (int r = 0; r < 8; ++r)
      sMid[r + hi8][nt * 16 + n] = sigmoidf_(acc[nt][r]);
  for (int idx = lane; idx < 16 * 16; idx += 32) sMid[idx >> 4][80 + (idx & 15)] = 0.0f;
  __syncthreads();

  // GEMM2: [16,80(+pad)]@[80,40]+b2 -> sigmoid (N padded to 48)
  v8f acc2[3];
#pragma unroll
  for (int nt = 0; nt < 3; ++nt) {
    int nn = nt * 16 + n;
    float bv = (nn < 40) ? b2[nn] : 0.0f;
#pragma unroll
    for (int r = 0; r < 8; ++r) acc2[nt][r] = bv;
  }
#pragma unroll
  for (int ks = 0; ks < 3; ++ks) {
    v16h a = load_a_lds(&sMid[0][0], 96, ks * 32, lane);
#pragma unroll
    for (int nt = 0; nt < 3; ++nt)
      acc2[nt] = wmma_f16(a, load_b_global(W2, 40, ks * 32, nt * 16, 80, 40, lane), acc2[nt]);
  }
  __syncthreads();
#pragma unroll
  for (int nt = 0; nt < 3; ++nt)
#pragma unroll
    for (int r = 0; r < 8; ++r)
      sMid[r + hi8][nt * 16 + n] = sigmoidf_(acc2[nt][r]);   // reuse sMid as [16][48]
  __syncthreads();

  // GEMM3: [16,40]@[40,1]+b3 (tiny: scalar per row)
  if (lane < 16) {
    float s = b3[0];
    for (int j = 0; j < 40; ++j) s += sMid[lane][j] * W3[j];
    score[(size_t)row0 + lane] = s;
  }
}

// ---------- kernel 4: masked softmax over T per batch row ----------
__global__ void softmax_kernel(const float* __restrict__ score, const unsigned char* __restrict__ mask,
                               float* __restrict__ att) {
  const int b = blockIdx.x, lane = threadIdx.x;
  float mx = -1e30f;
  for (int t = lane; t < TT; t += 32) {
    float s = mask[(size_t)b * TT + t] ? score[(size_t)b * TT + t] : -1e9f;
    mx = fmaxf(mx, s);
  }
#pragma unroll
  for (int off = 16; off; off >>= 1) mx = fmaxf(mx, __shfl_xor(mx, off, 32));
  float sum = 0.0f;
  for (int t = lane; t < TT; t += 32) {
    float s = mask[(size_t)b * TT + t] ? score[(size_t)b * TT + t] : -1e9f;
    float e = __expf(s - mx);
    att[(size_t)b * TT + t] = e;
    sum += e;
  }
#pragma unroll
  for (int off = 16; off; off >>= 1) sum += __shfl_xor(sum, off, 32);
  float inv = 1.0f / sum;
  for (int t = lane; t < TT; t += 32) att[(size_t)b * TT + t] *= inv;
}

// ---------- kernel 6: AUGRU scan -> final hidden state ----------
__global__ void augru_scan_kernel(const float* __restrict__ xp, const float* __restrict__ att,
                                  const unsigned char* __restrict__ mask,
                                  const float* __restrict__ Wh, float* __restrict__ hfinal) {
  __shared__ float sH[16][32];
  __shared__ float sT[16][32];
  __shared__ float sXP[2][16][96];
  const int lane = threadIdx.x & 31;
  const int b0 = blockIdx.x * 16;
  const int n = lane & 15, hi8 = (lane >> 4) << 3;

  v16h bu0 = load_b_global(Wh, 96, 0,  0, 32, 96, lane);
  v16h bu1 = load_b_global(Wh, 96, 0, 16, 32, 96, lane);
  v16h br0 = load_b_global(Wh, 96, 0, 32, 32, 96, lane);
  v16h br1 = load_b_global(Wh, 96, 0, 48, 32, 96, lane);
  v16h bh0 = load_b_global(Wh, 96, 0, 64, 32, 96, lane);
  v16h bh1 = load_b_global(Wh, 96, 0, 80, 32, 96, lane);

  v8f h_lo = {}, h_hi = {};
  for (int idx = lane; idx < 16 * 32; idx += 32) sH[idx >> 5][idx & 31] = 0.0f;
  __syncthreads();

  prefetch_xp(xp, b0, 0, &sXP[0][0][0]);

  for (int t = 0; t < TT; ++t) {
    if (t + 1 < TT) { prefetch_xp(xp, b0, t + 1, &sXP[(t + 1) & 1][0][0]); wait_xp_1(); }
    else            { wait_xp_0(); }
    const float (*xt)[96] = sXP[t & 1];

    v16h ah = load_a_lds(&sH[0][0], 32, 0, lane);
    v8f zero = {};
    v8f hu0 = wmma_f16(ah, bu0, zero);
    v8f hu1 = wmma_f16(ah, bu1, zero);
    v8f hr0 = wmma_f16(ah, br0, zero);
    v8f hr1 = wmma_f16(ah, br1, zero);

    v8f u_lo, u_hi, r_lo, r_hi;
#pragma unroll
    for (int r = 0; r < 8; ++r) {
      const float* xrow = xt[r + hi8];
      u_lo[r] = sigmoidf_(xrow[n]      + hu0[r]);
      u_hi[r] = sigmoidf_(xrow[16 + n] + hu1[r]);
      r_lo[r] = sigmoidf_(xrow[32 + n] + hr0[r]);
      r_hi[r] = sigmoidf_(xrow[48 + n] + hr1[r]);
      sT[r + hi8][n]      = r_lo[r] * h_lo[r];
      sT[r + hi8][16 + n] = r_hi[r] * h_hi[r];
    }
    __syncthreads();

    v16h arh = load_a_lds(&sT[0][0], 32, 0, lane);
    v8f hh0 = wmma_f16(arh, bh0, zero);
    v8f hh1 = wmma_f16(arh, bh1, zero);

#pragma unroll
    for (int r = 0; r < 8; ++r) {
      const int row = b0 + r + hi8;
      const float* xrow = xt[r + hi8];
      float a  = att[(size_t)row * TT + t];
      float hhl = tanhf(xrow[64 + n] + hh0[r]);
      float hhh = tanhf(xrow[80 + n] + hh1[r]);
      float ua0 = a * u_lo[r], ua1 = a * u_hi[r];
      float hn0 = (1.0f - ua0) * h_lo[r] + ua0 * hhl;
      float hn1 = (1.0f - ua1) * h_hi[r] + ua1 * hhh;
      bool mk = mask[(size_t)row * TT + t] != 0;
      h_lo[r] = mk ? hn0 : h_lo[r];
      h_hi[r] = mk ? hn1 : h_hi[r];
      sH[r + hi8][n]      = h_lo[r];
      sH[r + hi8][16 + n] = h_hi[r];
    }
    __syncthreads();
  }
#pragma unroll
  for (int r = 0; r < 8; ++r) {
    hfinal[(size_t)(b0 + r + hi8) * 32 + n]      = h_lo[r];
    hfinal[(size_t)(b0 + r + hi8) * 32 + 16 + n] = h_hi[r];
  }
}

// ---------- kernel 7: BN + FFN(PReLU) + out softmax ----------
__global__ void head_kernel(const float* __restrict__ hf, const float* __restrict__ tseq,
                            const float* __restrict__ tside, const float* __restrict__ uside,
                            const float* __restrict__ gamma, const float* __restrict__ beta,
                            const float* __restrict__ mean, const float* __restrict__ var,
                            const float* __restrict__ W1, const float* __restrict__ b1,
                            const float* __restrict__ a1,
                            const float* __restrict__ W2, const float* __restrict__ b2,
                            const float* __restrict__ a2,
                            const float* __restrict__ Wo, const float* __restrict__ bo,
                            float* __restrict__ out) {
  __shared__ float sIn[16][192];
  __shared__ float sX[16][224];
  __shared__ float sY[16][80];
  const int lane = threadIdx.x;
  const int b0 = blockIdx.x * 16;
  const int n = lane & 15, hi8 = (lane >> 4) << 3;

  for (int idx = lane; idx < 16 * 192; idx += 32) {
    int rr = idx / 192, c = idx % 192;
    int row = b0 + rr;
    float v;
    if (c < 32)       v = hf[(size_t)row * 32 + c];
    else if (c < 64)  v = tseq[(size_t)row * 32 + (c - 32)];
    else if (c < 128) v = tside[(size_t)row * 64 + (c - 64)];
    else              v = uside[(size_t)row * 64 + (c - 128)];
    sIn[rr][c] = (v - mean[c]) * rsqrtf(var[c] + EPS) * gamma[c] + beta[c];
  }
  __syncthreads();

  // GEMM1: [16,192]@[192,200]+b1 -> PReLU  (N padded to 208)
  v8f acc[13];
#pragma unroll
  for (int nt = 0; nt < 13; ++nt) {
    int nn = nt * 16 + n;
    float bv = (nn < 200) ? b1[nn] : 0.0f;
#pragma unroll
    for (int r = 0; r < 8; ++r) acc[nt][r] = bv;
  }
#pragma unroll
  for (int ks = 0; ks < 6; ++ks) {
    v16h a = load_a_lds(&sIn[0][0], 192, ks * 32, lane);
#pragma unroll
    for (int nt = 0; nt < 13; ++nt)
      acc[nt] = wmma_f16(a, load_b_global(W1, 200, ks * 32, nt * 16, 192, 200, lane), acc[nt]);
  }
#pragma unroll
  for (int nt = 0; nt < 13; ++nt) {
    int nn = nt * 16 + n;
    if (nn < 200) {
      float al = a1[nn];
#pragma unroll
      for (int r = 0; r < 8; ++r) {
        float x = acc[nt][r];
        sX[r + hi8][nn] = (x >= 0.0f) ? x : al * x;
      }
    }
  }
  for (int idx = lane; idx < 16 * 24; idx += 32) sX[idx / 24][200 + idx % 24] = 0.0f;
  __syncthreads();

  // GEMM2: [16,200(+pad)]@[200,80]+b2 -> PReLU
  v8f acc2[5];
#pragma unroll
  for (int nt = 0; nt < 5; ++nt) {
    float bv = b2[nt * 16 + n];
#pragma unroll
    for (int r = 0; r < 8; ++r) acc2[nt][r] = bv;
  }
#pragma unroll
  for (int ks = 0; ks < 7; ++ks) {
    v16h a = load_a_lds(&sX[0][0], 224, ks * 32, lane);
#pragma unroll
    for (int nt = 0; nt < 5; ++nt)
      acc2[nt] = wmma_f16(a, load_b_global(W2, 80, ks * 32, nt * 16, 200, 80, lane), acc2[nt]);
  }
#pragma unroll
  for (int nt = 0; nt < 5; ++nt) {
    int nn = nt * 16 + n;
    float al = a2[nn];
#pragma unroll
    for (int r = 0; r < 8; ++r) {
      float x = acc2[nt][r];
      sY[r + hi8][nn] = (x >= 0.0f) ? x : al * x;
    }
  }
  __syncthreads();

  // out: [16,80]@[80,2]+bo -> softmax(2)
  if (lane < 16) {
    float l0 = bo[0], l1 = bo[1];
    for (int j = 0; j < 80; ++j) {
      float v = sY[lane][j];
      l0 += v * Wo[j * 2 + 0];
      l1 += v * Wo[j * 2 + 1];
    }
    float mx = fmaxf(l0, l1);
    float e0 = __expf(l0 - mx), e1 = __expf(l1 - mx);
    float inv = 1.0f / (e0 + e1);
    out[(size_t)(b0 + lane) * 2 + 0] = e0 * inv;
    out[(size_t)(b0 + lane) * 2 + 1] = e1 * inv;
  }
}

// ---------- launch ----------
extern "C" void kernel_launch(void* const* d_in, const int* in_sizes, int n_in,
                              void* d_out, int out_size, void* d_ws, size_t ws_size,
                              hipStream_t stream) {
  const unsigned char* mask = (const unsigned char*)d_in[0];
  const float* user_side    = (const float*)d_in[1];
  const float* seq_embed    = (const float*)d_in[2];
  const float* tgt_seq      = (const float*)d_in[3];
  const float* tgt_side     = (const float*)d_in[4];
  const float* gru_Wx       = (const float*)d_in[5];
  const float* gru_Wh       = (const float*)d_in[6];
  const float* gru_b        = (const float*)d_in[7];
  const float* att_W1       = (const float*)d_in[8];
  const float* att_b1       = (const float*)d_in[9];
  const float* att_W2       = (const float*)d_in[10];
  const float* att_b2       = (const float*)d_in[11];
  const float* att_W3       = (const float*)d_in[12];
  const float* att_b3       = (const float*)d_in[13];
  const float* aug_Wx       = (const float*)d_in[14];
  const float* aug_Wh       = (const float*)d_in[15];
  const float* aug_b        = (const float*)d_in[16];
  const float* bn_gamma     = (const float*)d_in[17];
  const float* bn_beta      = (const float*)d_in[18];
  const float* bn_mean      = (const float*)d_in[19];
  const float* bn_var       = (const float*)d_in[20];
  const float* ffn_W1       = (const float*)d_in[21];
  const float* ffn_b1       = (const float*)d_in[22];
  const float* ffn_a1       = (const float*)d_in[23];
  const float* ffn_W2       = (const float*)d_in[24];
  const float* ffn_b2       = (const float*)d_in[25];
  const float* ffn_a2       = (const float*)d_in[26];
  const float* out_W        = (const float*)d_in[27];
  const float* out_b        = (const float*)d_in[28];
  float* out = (float*)d_out;

  char* ws = (char*)d_ws;
  const size_t M = (size_t)BB * TT;                  // 819200 flattened rows
  float* xp    = (float*)(ws);                       // [B*T,96] (reused for xp2)
  float* ge    = (float*)(ws + M * 96 * 4);          // [B*T,32]
  float* score = (float*)(ws + M * 96 * 4 + M * 32 * 4);
  float* att   = score + M;
  float* hfin  = att + M;

  // 1) GRU input projection: xp = seq_embed @ gru_Wx + gru_b
  proj_kernel<<<(int)(M / 16) / PROJ_WPB, 32 * PROJ_WPB, 0, stream>>>(
      seq_embed, gru_Wx, gru_b, xp, (int)M);
  // 2) GRU recurrence (TDM double-buffered xp streaming)
  gru_scan_kernel<<<BB / 16, 32, 0, stream>>>(xp, mask, gru_Wh, ge);
  // 3) DIN attention MLP -> raw score
  attn_kernel<<<(int)(M / 16) / ATT_WPB, 32 * ATT_WPB, 0, stream>>>(
      tgt_seq, ge, att_W1, att_b1, att_W2, att_b2, att_W3, att_b3, score);
  // 4) masked softmax over T
  softmax_kernel<<<BB, 32, 0, stream>>>(score, mask, att);
  // 5) AUGRU input projection: xp2 = ge @ aug_Wx + aug_b (reuse xp buffer)
  proj_kernel<<<(int)(M / 16) / PROJ_WPB, 32 * PROJ_WPB, 0, stream>>>(
      ge, aug_Wx, aug_b, xp, (int)M);
  // 6) AUGRU recurrence -> h_final
  augru_scan_kernel<<<BB / 16, 32, 0, stream>>>(xp, att, mask, aug_Wh, hfin);
  // 7) BN + FFN + output softmax
  head_kernel<<<BB / 16, 32, 0, stream>>>(hfin, tgt_seq, tgt_side, user_side,
                                          bn_gamma, bn_beta, bn_mean, bn_var,
                                          ffn_W1, ffn_b1, ffn_a1,
                                          ffn_W2, ffn_b2, ffn_a2,
                                          out_W, out_b, out);
}
